// DimNetInteractionPPBlock_24953759989850
// MI455X (gfx1250) — matrix operand order repaired and checked by reading, not compiled, not run on hardware
//
#include <hip/hip_runtime.h>
#include <hip/hip_bf16.h>

typedef __bf16 bf16_t;
typedef __attribute__((ext_vector_type(16))) __bf16 v16bf;
typedef __attribute__((ext_vector_type(8)))  __bf16 v8bf;
typedef __attribute__((ext_vector_type(8)))  float  v8f;

#define DEV __device__ __forceinline__

DEV float silu_f(float v) { return v / (1.0f + __expf(-v)); }

// ---------------------------------------------------------------------------
// WMMA fragment helpers (wave32, 16x16x32 bf16 -> f32).
// A layout: lane m = lane&15; element e -> K = e + 16*(lane>=16)
// B layout: lane n = lane&15; element e -> K = e + 16*(lane>=16)
// C layout: reg r -> M = r + 8*(lane>=16); N = lane&15
// ---------------------------------------------------------------------------

// Load an A fragment (16 rows x 32 K) from a row-major f32 global matrix.
DEV v16bf load_A_gf32(const float* __restrict__ base, int ld, int row0, int k0, int lane) {
  const int m    = lane & 15;
  const int koff = (lane >> 4) << 4;
  const float* p = base + (size_t)(row0 + m) * ld + k0 + koff;
  float4 a0 = *(const float4*)(p + 0);
  float4 a1 = *(const float4*)(p + 4);
  float4 b0 = *(const float4*)(p + 8);
  float4 b1 = *(const float4*)(p + 12);
  v16bf o;
  o[0]=(bf16_t)a0.x; o[1]=(bf16_t)a0.y; o[2]=(bf16_t)a0.z;  o[3]=(bf16_t)a0.w;
  o[4]=(bf16_t)a1.x; o[5]=(bf16_t)a1.y; o[6]=(bf16_t)a1.z;  o[7]=(bf16_t)a1.w;
  o[8]=(bf16_t)b0.x; o[9]=(bf16_t)b0.y; o[10]=(bf16_t)b0.z; o[11]=(bf16_t)b0.w;
  o[12]=(bf16_t)b1.x;o[13]=(bf16_t)b1.y;o[14]=(bf16_t)b1.z; o[15]=(bf16_t)b1.w;
  return o;
}

// Load an A fragment from per-wave LDS staging buffer, row-major bf16 [16][128].
DEV v16bf load_A_lds(const bf16_t* st, int k0, int lane) {
  const int m    = lane & 15;
  const int koff = (lane >> 4) << 4;
  const bf16_t* p = st + m * 128 + k0 + koff;
  v8bf lo = *(const v8bf*)(p);
  v8bf hi = *(const v8bf*)(p + 8);
  v16bf o;
#pragma unroll
  for (int e = 0; e < 8; ++e) { o[e] = lo[e]; o[e + 8] = hi[e]; }
  return o;
}

// GEMM row-tile: C[16,16] += A[16,32*KT] x W[32*KT,16] using pre-swizzled
// weights: tile (kt,nt) occupies 512 bf16; lane's 16 values are contiguous.
template <int KT>
DEV v8f gemm_tile(const v16bf* a, const bf16_t* w, int nt, int NT, int lane) {
  v8f c = {0.f, 0.f, 0.f, 0.f, 0.f, 0.f, 0.f, 0.f};
#pragma unroll
  for (int kt = 0; kt < KT; ++kt) {
    v16bf b = *(const v16bf*)(w + (((size_t)(kt * NT + nt)) << 9) + (lane << 4));
    c = __builtin_amdgcn_wmma_f32_16x16x32_bf16(false, a[kt], false, b,
                                                (short)0, c, false, false);
  }
  return c;
}

// ---------------------------------------------------------------------------
// Weight packing: f32 row-major [Kd,Nd] -> bf16 WMMA-B-swizzled blob.
// dst[idx]: idx = tile*512 + lane*16 + e ; tile = kt*(Nd/16)+nt
// source (k,n): k = kt*32 + 16*(lane>=16) + e ; n = nt*16 + (lane&15)
// ---------------------------------------------------------------------------
__global__ void k_pack_weight(const float* __restrict__ W, bf16_t* __restrict__ dst,
                              int Kd, int Nd) {
  int idx = blockIdx.x * 256 + threadIdx.x;
  if (idx >= Kd * Nd) return;
  int t = idx >> 9, r = idx & 511, L = r >> 4, e = r & 15;
  int NT = Nd >> 4;
  int kt = t / NT, nt = t % NT;
  int k = kt * 32 + ((L >> 4) << 4) + e;
  int n = nt * 16 + (L & 15);
  dst[idx] = (bf16_t)W[k * Nd + n];
}

// Packs (W1 [Kd,8] @ W2 [8,Nd]) directly into the swizzled blob.
__global__ void k_pack_combo(const float* __restrict__ W1, const float* __restrict__ W2,
                             bf16_t* __restrict__ dst, int Kd, int Nd) {
  int idx = blockIdx.x * 256 + threadIdx.x;
  if (idx >= Kd * Nd) return;
  int t = idx >> 9, r = idx & 511, L = r >> 4, e = r & 15;
  int NT = Nd >> 4;
  int kt = t / NT, nt = t % NT;
  int k = kt * 32 + ((L >> 4) << 4) + e;
  int n = nt * 16 + (L & 15);
  float s = 0.f;
#pragma unroll
  for (int q = 0; q < 8; ++q) s += W1[k * 8 + q] * W2[q * Nd + n];
  dst[idx] = (bf16_t)s;
}

__global__ void k_zero_f4(float4* __restrict__ p, int n4) {
  int i = blockIdx.x * 256 + threadIdx.x;
  if (i < n4) p[i] = make_float4(0.f, 0.f, 0.f, 0.f);
}

// ---------------------------------------------------------------------------
// Kernel 1: per-edge pre.
//   x_ji = silu(x@W_ji + b_ji)                       -> ws (f32, [M,128])
//   h    = silu((silu(x@W_kj+b_kj) * (rbf@Wrbf)) @ W_down)  -> ws (bf16,[M,64])
// Blob slice: [W_ji 16384 | W_kj 16384 | Wrbf 8192 | W_down 8192]
// ---------------------------------------------------------------------------
__global__ __launch_bounds__(256) void k_edge_pre(
    const float* __restrict__ x, const float* __restrict__ rbf,
    const float* __restrict__ b_ji, const float* __restrict__ b_kj,
    const bf16_t* __restrict__ wblob, float* __restrict__ xji,
    bf16_t* __restrict__ hbuf, int M) {
  __shared__ __align__(32) bf16_t sW[49152];
  __shared__ __align__(32) bf16_t sStage[8 * 16 * 128];
  {
    const int4* src = (const int4*)wblob;
    int4* dst = (int4*)sW;
    for (int i = threadIdx.x; i < 6144; i += 256) dst[i] = src[i];
  }
  __syncthreads();
  const int wave = threadIdx.x >> 5, lane = threadIdx.x & 31;
  const int tile = blockIdx.x * 8 + wave;
  if (tile * 16 >= M) return;
  const int row0 = tile * 16;
  bf16_t* st = sStage + wave * (16 * 128);
  const int n = lane & 15;
  const int moff = (lane >> 4) << 3;

  v16bf ax[4];
#pragma unroll
  for (int kt = 0; kt < 4; ++kt) ax[kt] = load_A_gf32(x, 128, row0, kt * 32, lane);
  v16bf ar[2];
#pragma unroll
  for (int kt = 0; kt < 2; ++kt) ar[kt] = load_A_gf32(rbf, 64, row0, kt * 32, lane);

  const bf16_t* wJI = sW;
  const bf16_t* wKJ = sW + 16384;
  const bf16_t* wRB = sW + 32768;
  const bf16_t* wDN = sW + 40960;

  // x_ji = silu(x@W_ji + b_ji) -> global (C-layout maps to row-major)
#pragma unroll
  for (int nt = 0; nt < 8; ++nt) {
    v8f c = gemm_tile<4>(ax, wJI, nt, 8, lane);
    float b = b_ji[nt * 16 + n];
#pragma unroll
    for (int r = 0; r < 8; ++r)
      xji[(size_t)(row0 + moff + r) * 128 + nt * 16 + n] = silu_f(c[r] + b);
  }
  // t = silu(x@W_kj+b_kj) * (rbf@Wrbf)  -> stage bf16
#pragma unroll
  for (int nt = 0; nt < 8; ++nt) {
    v8f c  = gemm_tile<4>(ax, wKJ, nt, 8, lane);
    v8f rt = gemm_tile<2>(ar, wRB, nt, 8, lane);
    float b = b_kj[nt * 16 + n];
#pragma unroll
    for (int r = 0; r < 8; ++r)
      st[(moff + r) * 128 + nt * 16 + n] = (bf16_t)(silu_f(c[r] + b) * rt[r]);
  }
  // h = silu(t @ W_down) -> global bf16 [M,64] (L2-resident gather source)
  v16bf ah[4];
#pragma unroll
  for (int kt = 0; kt < 4; ++kt) ah[kt] = load_A_lds(st, kt * 32, lane);
#pragma unroll
  for (int nt = 0; nt < 4; ++nt) {
    v8f c = gemm_tile<4>(ah, wDN, nt, 4, lane);
#pragma unroll
    for (int r = 0; r < 8; ++r)
      hbuf[(size_t)(row0 + moff + r) * 64 + nt * 16 + n] = (bf16_t)silu_f(c[r]);
  }
}

// ---------------------------------------------------------------------------
// Kernel 2: per-triplet. sbf_t = sbf@Wsbf (WMMA) -> LDS; gather h[idx_j] (bf16,
// L2-resident), multiply, fp32 atomic scatter-add into acc[idx_i].
// ---------------------------------------------------------------------------
__global__ __launch_bounds__(256) void k_triplet(
    const float* __restrict__ sbf, const bf16_t* __restrict__ hbuf,
    const int* __restrict__ aidx, const bf16_t* __restrict__ wsbf,
    float* __restrict__ acc, int K) {
  __shared__ __align__(32) bf16_t sW[4096];
  __shared__ float sT[128 * 64];
  {
    const int4* src = (const int4*)wsbf;
    int4* dst = (int4*)sW;
    for (int i = threadIdx.x; i < 512; i += 256) dst[i] = src[i];
  }
  __syncthreads();
  const int wave = threadIdx.x >> 5, lane = threadIdx.x & 31;
  const int row0 = blockIdx.x * 128 + wave * 16;
  if (row0 >= K) return;
  float* st = sT + wave * (16 * 64);
  const int n = lane & 15, moff = (lane >> 4) << 3;

  v16bf as[2];
#pragma unroll
  for (int kt = 0; kt < 2; ++kt) as[kt] = load_A_gf32(sbf, 64, row0, kt * 32, lane);
#pragma unroll
  for (int nt = 0; nt < 4; ++nt) {
    v8f c = gemm_tile<2>(as, sW, nt, 4, lane);
#pragma unroll
    for (int r = 0; r < 8; ++r) st[(moff + r) * 64 + nt * 16 + n] = c[r];
  }
  // gather + scatter: lane -> (local row = lane>>1, 32 cols = (lane&1)*32)
  const int lr = lane >> 1;
  const int gr = row0 + lr;
  const int c0 = (lane & 1) << 5;
  const int j = aidx[(size_t)K + gr];  // angle_index[1] : gather source
  const int i = aidx[gr];              // angle_index[0] : segment id
  const bf16_t* hp = hbuf + (size_t)j * 64 + c0;
  float* ap = acc + (size_t)i * 64 + c0;
  const float* sp = st + lr * 64 + c0;
#pragma unroll
  for (int q = 0; q < 4; ++q) {
    v8bf hv = *(const v8bf*)(hp + q * 8);
#pragma unroll
    for (int e = 0; e < 8; ++e) {
      float v = (float)hv[e] * sp[q * 8 + e];
      __hip_atomic_fetch_add(ap + q * 8 + e, v, __ATOMIC_RELAXED,
                             __HIP_MEMORY_SCOPE_AGENT);
    }
  }
}

// ---------------------------------------------------------------------------
// Kernel 3a: x2 = x_ji + silu(acc@W_up); x2 += silu(silu(x2@Wb1+bb1)@Wb2+bb2);
//            mid = x + silu(x2@W_fin+b_fin)   (mid overwrites x_ji buffer)
// Blob slice: [W_up 8192 | Wb1 16384 | Wb2 16384 | W_fin 16384]
// ---------------------------------------------------------------------------
__global__ __launch_bounds__(256) void k_edge_post1(
    const float* __restrict__ acc, const float* __restrict__ x,
    const float* __restrict__ bb1, const float* __restrict__ bb2,
    const float* __restrict__ bfin, const bf16_t* __restrict__ wblob,
    float* __restrict__ xji_mid, int M) {
  __shared__ __align__(32) bf16_t sW[57344];
  __shared__ __align__(32) bf16_t sA[8 * 16 * 128];
  __shared__ float sR[8 * 16 * 128];
  {
    const int4* src = (const int4*)wblob;
    int4* dst = (int4*)sW;
    for (int i = threadIdx.x; i < 7168; i += 256) dst[i] = src[i];
  }
  __syncthreads();
  const int wave = threadIdx.x >> 5, lane = threadIdx.x & 31;
  const int tile = blockIdx.x * 8 + wave;
  if (tile * 16 >= M) return;
  const int row0 = tile * 16;
  bf16_t* stA = sA + wave * 2048;
  float*  stR = sR + wave * 2048;
  const int n = lane & 15, moff = (lane >> 4) << 3;
  const bf16_t* wUP = sW;
  const bf16_t* wB1 = sW + 8192;
  const bf16_t* wB2 = sW + 24576;
  const bf16_t* wFN = sW + 40960;

  v16bf aa[2];
#pragma unroll
  for (int kt = 0; kt < 2; ++kt) aa[kt] = load_A_gf32(acc, 64, row0, kt * 32, lane);
  // x2 = x_ji + silu(acc@W_up)   (W_up has no bias in reference)
#pragma unroll
  for (int nt = 0; nt < 8; ++nt) {
    v8f c = gemm_tile<2>(aa, wUP, nt, 8, lane);
#pragma unroll
    for (int r = 0; r < 8; ++r) {
      float v = xji_mid[(size_t)(row0 + moff + r) * 128 + nt * 16 + n] + silu_f(c[r]);
      stA[(moff + r) * 128 + nt * 16 + n] = (bf16_t)v;
      stR[(moff + r) * 128 + nt * 16 + n] = v;
    }
  }
  // t1 = silu(x2@Wb1+bb1)
  v16bf a2[4];
#pragma unroll
  for (int kt = 0; kt < 4; ++kt) a2[kt] = load_A_lds(stA, kt * 32, lane);
#pragma unroll
  for (int nt = 0; nt < 8; ++nt) {
    v8f c = gemm_tile<4>(a2, wB1, nt, 8, lane);
    float b = bb1[nt * 16 + n];
#pragma unroll
    for (int r = 0; r < 8; ++r)
      stA[(moff + r) * 128 + nt * 16 + n] = (bf16_t)silu_f(c[r] + b);
  }
  // x2' = x2 + silu(t1@Wb2+bb2)
  v16bf a1[4];
#pragma unroll
  for (int kt = 0; kt < 4; ++kt) a1[kt] = load_A_lds(stA, kt * 32, lane);
#pragma unroll
  for (int nt = 0; nt < 8; ++nt) {
    v8f c = gemm_tile<4>(a1, wB2, nt, 8, lane);
    float b = bb2[nt * 16 + n];
#pragma unroll
    for (int r = 0; r < 8; ++r) {
      float v = stR[(moff + r) * 128 + nt * 16 + n] + silu_f(c[r] + b);
      stA[(moff + r) * 128 + nt * 16 + n] = (bf16_t)v;
    }
  }
  // mid = x + silu(x2'@W_fin+b_fin)  (overwrite x_ji buffer in place)
  v16bf a3[4];
#pragma unroll
  for (int kt = 0; kt < 4; ++kt) a3[kt] = load_A_lds(stA, kt * 32, lane);
#pragma unroll
  for (int nt = 0; nt < 8; ++nt) {
    v8f c = gemm_tile<4>(a3, wFN, nt, 8, lane);
    float b = bfin[nt * 16 + n];
#pragma unroll
    for (int r = 0; r < 8; ++r) {
      float v = x[(size_t)(row0 + moff + r) * 128 + nt * 16 + n] + silu_f(c[r] + b);
      xji_mid[(size_t)(row0 + moff + r) * 128 + nt * 16 + n] = v;
    }
  }
}

// ---------------------------------------------------------------------------
// Kernel 3b: two after-skip residual blocks:
//   out = mid; out = out + silu(silu(out@Wa1[i]+ba1[i])@Wa2[i]+ba2[i]), i=0,1
// Blob slice: [Wa1_0 | Wa2_0 | Wa1_1 | Wa2_1] = 65536 bf16
// ---------------------------------------------------------------------------
__global__ __launch_bounds__(256) void k_edge_post2(
    const float* __restrict__ mid, const float* __restrict__ ba1,
    const float* __restrict__ ba2, const bf16_t* __restrict__ wblob,
    float* __restrict__ out, int M) {
  __shared__ __align__(32) bf16_t sW[65536];
  __shared__ __align__(32) bf16_t sA[8 * 16 * 128];
  __shared__ float sR[8 * 16 * 128];
  {
    const int4* src = (const int4*)wblob;
    int4* dst = (int4*)sW;
    for (int i = threadIdx.x; i < 8192; i += 256) dst[i] = src[i];
  }
  __syncthreads();
  const int wave = threadIdx.x >> 5, lane = threadIdx.x & 31;
  const int tile = blockIdx.x * 8 + wave;
  if (tile * 16 >= M) return;
  const int row0 = tile * 16;
  bf16_t* stA = sA + wave * 2048;
  float*  stR = sR + wave * 2048;
  const int n = lane & 15, moff = (lane >> 4) << 3;

#pragma unroll
  for (int nt = 0; nt < 8; ++nt) {
#pragma unroll
    for (int r = 0; r < 8; ++r) {
      float v = mid[(size_t)(row0 + moff + r) * 128 + nt * 16 + n];
      stR[(moff + r) * 128 + nt * 16 + n] = v;
      stA[(moff + r) * 128 + nt * 16 + n] = (bf16_t)v;
    }
  }
#pragma unroll
  for (int blk = 0; blk < 2; ++blk) {
    const bf16_t* Wp = sW + blk * 32768;
    const bf16_t* Wq = Wp + 16384;
    const float* bp = ba1 + blk * 128;
    const float* bq = ba2 + blk * 128;
    v16bf ac[4];
#pragma unroll
    for (int kt = 0; kt < 4; ++kt) ac[kt] = load_A_lds(stA, kt * 32, lane);
#pragma unroll
    for (int nt = 0; nt < 8; ++nt) {
      v8f c = gemm_tile<4>(ac, Wp, nt, 8, lane);
      float b = bp[nt * 16 + n];
#pragma unroll
      for (int r = 0; r < 8; ++r)
        stA[(moff + r) * 128 + nt * 16 + n] = (bf16_t)silu_f(c[r] + b);
    }
    v16bf at[4];
#pragma unroll
    for (int kt = 0; kt < 4; ++kt) at[kt] = load_A_lds(stA, kt * 32, lane);
#pragma unroll
    for (int nt = 0; nt < 8; ++nt) {
      v8f c = gemm_tile<4>(at, Wq, nt, 8, lane);
      float b = bq[nt * 16 + n];
#pragma unroll
      for (int r = 0; r < 8; ++r) {
        float v = stR[(moff + r) * 128 + nt * 16 + n] + silu_f(c[r] + b);
        stR[(moff + r) * 128 + nt * 16 + n] = v;
        stA[(moff + r) * 128 + nt * 16 + n] = (bf16_t)v;
      }
    }
  }
#pragma unroll
  for (int nt = 0; nt < 8; ++nt)
#pragma unroll
    for (int r = 0; r < 8; ++r)
      out[(size_t)(row0 + moff + r) * 128 + nt * 16 + n] =
          stR[(moff + r) * 128 + nt * 16 + n];
}

// ---------------------------------------------------------------------------
// Host launcher
// ---------------------------------------------------------------------------
extern "C" void kernel_launch(void* const* d_in, const int* in_sizes, int n_in,
                              void* d_out, int out_size, void* d_ws, size_t ws_size,
                              hipStream_t stream) {
  const float* x      = (const float*)d_in[0];
  const float* rbf    = (const float*)d_in[1];
  const float* sbf    = (const float*)d_in[2];
  const int*   aidx   = (const int*)d_in[3];
  const float* W_rbf1 = (const float*)d_in[4];
  const float* W_rbf2 = (const float*)d_in[5];
  const float* W_sbf1 = (const float*)d_in[6];
  const float* W_sbf2 = (const float*)d_in[7];
  const float* W_ji   = (const float*)d_in[8];
  const float* b_ji   = (const float*)d_in[9];
  const float* W_kj   = (const float*)d_in[10];
  const float* b_kj   = (const float*)d_in[11];
  const float* W_down = (const float*)d_in[12];
  const float* W_up   = (const float*)d_in[13];
  const float* Wb1    = (const float*)d_in[14];
  const float* bb1    = (const float*)d_in[15];
  const float* Wb2    = (const float*)d_in[16];
  const float* bb2    = (const float*)d_in[17];
  const float* W_fin  = (const float*)d_in[18];
  const float* b_fin  = (const float*)d_in[19];
  const float* Wa1    = (const float*)d_in[20];
  const float* ba1    = (const float*)d_in[21];
  const float* Wa2    = (const float*)d_in[22];
  const float* ba2    = (const float*)d_in[23];

  const int M = in_sizes[0] / 128;
  const int K = in_sizes[2] / 64;

  // ws layout: [bf16 weight blob | acc f32 M*64 | xji/mid f32 M*128 | h bf16 M*64]
  bf16_t* wsW  = (bf16_t*)d_ws;
  char*   base = (char*)d_ws;
  const size_t ACC_OFF = (size_t)1 << 20;
  float*  acc  = (float*)(base + ACC_OFF);
  float*  xji  = (float*)(base + ACC_OFF + (size_t)M * 64 * 4);
  bf16_t* hbuf = (bf16_t*)(base + ACC_OFF + (size_t)M * 64 * 4 + (size_t)M * 128 * 4);

  // 0) zero the segment-sum accumulator (must happen every call)
  int n4 = M * 16;  // M*64/4 float4s
  k_zero_f4<<<dim3((n4 + 255) / 256), dim3(256), 0, stream>>>((float4*)acc, n4);

  // 0b) pack all weights into WMMA-swizzled bf16 blob
  auto packW = [&](const float* W, int off, int Kd, int Nd) {
    k_pack_weight<<<dim3((Kd * Nd + 255) / 256), dim3(256), 0, stream>>>(
        W, wsW + off, Kd, Nd);
  };
  packW(W_ji, 0, 128, 128);
  packW(W_kj, 16384, 128, 128);
  k_pack_combo<<<dim3((64 * 128 + 255) / 256), dim3(256), 0, stream>>>(
      W_rbf1, W_rbf2, wsW + 32768, 64, 128);
  packW(W_down, 40960, 128, 64);
  k_pack_combo<<<dim3((64 * 64 + 255) / 256), dim3(256), 0, stream>>>(
      W_sbf1, W_sbf2, wsW + 49152, 64, 64);
  packW(W_up, 53248, 64, 128);
  packW(Wb1, 61440, 128, 128);
  packW(Wb2, 77824, 128, 128);
  packW(W_fin, 94208, 128, 128);
  packW(Wa1, 110592, 128, 128);
  packW(Wa2, 126976, 128, 128);
  packW(Wa1 + 16384, 143360, 128, 128);
  packW(Wa2 + 16384, 159744, 128, 128);

  const int tilesM = (M + 15) / 16;
  const int gridM  = (tilesM + 7) / 8;
  const int gridK  = (K + 127) / 128;

  k_edge_pre<<<gridM, 256, 0, stream>>>(x, rbf, b_ji, b_kj, wsW, xji, hbuf, M);
  k_triplet<<<gridK, 256, 0, stream>>>(sbf, hbuf, aidx, wsW + 49152, acc, K);
  k_edge_post1<<<gridM, 256, 0, stream>>>(acc, x, bb1, bb2, b_fin, wsW + 53248, xji, M);
  k_edge_post2<<<gridM, 256, 0, stream>>>(xji, ba1, ba2, wsW + 110592,
                                          (float*)d_out, M);
}
